// SelfAttentionHead_4707284156974
// MI455X (gfx1250) — matrix operand order repaired and checked by reading, not compiled, not run on hardware
//
#include <hip/hip_runtime.h>
#include <hip/hip_bf16.h>

typedef __attribute__((ext_vector_type(16))) _Float16 v16h;
typedef __attribute__((ext_vector_type(8)))  _Float16 v8h;
typedef __attribute__((ext_vector_type(8)))  float    v8f;
typedef __attribute__((ext_vector_type(4)))  int      v4i;

// Problem dimensions (fixed by the reference)
constexpr int Bn = 8, Tn = 2048, Cn = 1024, Hn = 64;
constexpr int Nrow = Bn * Tn;                 // 16384 total rows
constexpr float kLog2e = 1.4426950408889634f;

__device__ __forceinline__ int laneid() { return (int)(threadIdx.x & 31u); }

// ---------------------------------------------------------------------------
// Async global -> LDS copy of one 16-byte segment (CDNA5 ASYNCcnt path).
// Builtin signature (from hipcc diagnostic): (int4 AS1*, int4 AS3*, Ii, Ii).
__device__ __forceinline__ void async_copy_b128(const _Float16* g, _Float16* l) {
#if __has_builtin(__builtin_amdgcn_global_load_async_to_lds_b128)
  __builtin_amdgcn_global_load_async_to_lds_b128(
      (__attribute__((address_space(1))) v4i*)g,
      (__attribute__((address_space(3))) v4i*)l, 0, 0);
#else
  unsigned lofs = (unsigned)(unsigned long long)l;  // LDS offset = low 32 bits
  asm volatile("global_load_async_to_lds_b128 %0, %1, off"
               :: "v"(lofs), "v"(g) : "memory");
#endif
}

__device__ __forceinline__ void wait_async0() {
#if __has_builtin(__builtin_amdgcn_s_wait_asynccnt)
  __builtin_amdgcn_s_wait_asynccnt(0);
#else
  asm volatile("s_wait_asynccnt 0" ::: "memory");
#endif
}

// ---------------------------------------------------------------------------
// WMMA fragment loaders (wave32, v_wmma_f32_16x16x32_f16 layouts per CDNA5 ISA)
//
// A (16x32, M x K), row-major source with leading dim `ld` (halves):
//   lane l: row m = l&15 ; lanes 0-15 hold K {0..7, 16..23}, lanes 16-31 hold
//   K {8..15, 24..31}  -> two 16B loads per lane.
__device__ __forceinline__ v16h load_fragA(const _Float16* base, int ld, int row0, int k0) {
  const int l  = laneid();
  const int m  = l & 15;
  const int kb = (l >> 4) * 8;
  const _Float16* p = base + (size_t)(row0 + m) * ld + (k0 + kb);
  v8h lo = *(const v8h*)(p);
  v8h hi = *(const v8h*)(p + 16);
  v16h out;
#pragma unroll
  for (int i = 0; i < 8; ++i) { out[i] = lo[i]; out[i + 8] = hi[i]; }
  return out;
}

// B (32x16, K x N) given B^T stored row-major: bt[n][k] with leading dim `ld`.
//   lane l: column n = l&15 ; lanes 0-15 hold K {0..7, 16..23}, lanes 16-31 hold K {8..15, 24..31}.
__device__ __forceinline__ v16h load_fragB(const _Float16* bt, int ld, int n0, int k0) {
  const int l  = laneid();
  const int n  = l & 15;
  const int kb = (l >> 4) * 8;
  const _Float16* p = bt + (size_t)(n0 + n) * ld + (k0 + kb);
  v8h lo = *(const v8h*)(p);
  v8h hi = *(const v8h*)(p + 16);
  v16h out;
#pragma unroll
  for (int i = 0; i < 8; ++i) { out[i] = lo[i]; out[i + 8] = hi[i]; }
  return out;
}

// A-fragment from a 16x32 f16 tile in LDS (row-major, ld = 32 halves).
__device__ __forceinline__ v16h load_fragA_lds(const _Float16* tile) {
  const int l  = laneid();
  const int m  = l & 15;
  const int kb = (l >> 4) * 8;
  const _Float16* p = tile + m * 32 + kb;
  v8h lo = *(const v8h*)(p);
  v8h hi = *(const v8h*)(p + 16);
  v16h out;
#pragma unroll
  for (int i = 0; i < 8; ++i) { out[i] = lo[i]; out[i + 8] = hi[i]; }
  return out;
}

// ---------------------------------------------------------------------------
// Kernel 0a: x fp32 -> f16
__global__ void __launch_bounds__(256)
cvt_x_kernel(const float* __restrict__ x, _Float16* __restrict__ xh, size_t nElem) {
  size_t i = (size_t)blockIdx.x * blockDim.x + threadIdx.x;
  if (i < nElem) xh[i] = (_Float16)x[i];
}

// Kernel 0b: Wq/Wk/Wv (C x H fp32) -> wt (3 x H x C f16, transposed)
__global__ void __launch_bounds__(256)
cvt_w_kernel(const float* __restrict__ Wq, const float* __restrict__ Wk,
             const float* __restrict__ Wv, _Float16* __restrict__ wt) {
  int i = blockIdx.x * blockDim.x + threadIdx.x;
  if (i >= 3 * Hn * Cn) return;
  int mat = i / (Hn * Cn);
  int rem = i % (Hn * Cn);
  int h = rem / Cn;
  int c = rem % Cn;
  const float* W = (mat == 0) ? Wq : (mat == 1) ? Wk : Wv;
  wt[i] = (_Float16)W[(size_t)c * Hn + h];
}

// ---------------------------------------------------------------------------
// Kernel 1: QKV projection. One wave per (16-row tile, matrix).
//   Q written * C^-0.5 (score scale folded in), row-major f16 [Nrow][H]
//   K row-major f16 [Nrow][H]
//   V written transposed per batch: vT[b][h][t], f16
__global__ void __launch_bounds__(32)
proj_kernel(const _Float16* __restrict__ xh, const _Float16* __restrict__ wt,
            _Float16* __restrict__ q16, _Float16* __restrict__ k16,
            _Float16* __restrict__ vTh) {
  const int row0 = blockIdx.x * 16;                 // global row
  const int mat  = blockIdx.y;                      // 0=q 1=k 2=v
  const _Float16* w = wt + (size_t)mat * Hn * Cn;   // [H][C] row-major (= W^T)

  v8f acc[4] = {};
  for (int k0 = 0; k0 < Cn; k0 += 32) {
    v16h a = load_fragA(xh, Cn, row0, k0);
#pragma unroll
    for (int nt = 0; nt < 4; ++nt) {
      v16h bw = load_fragB(w, Cn, nt * 16, k0);
      acc[nt] = __builtin_amdgcn_wmma_f32_16x16x32_f16(
          false, a, false, bw, (short)0, acc[nt], false, false);
    }
  }

  const int l   = laneid();
  const int n   = l & 15;
  const int hi8 = ((l >> 4) & 1) * 8;               // C/D: M = r + hi8, N = n
  if (mat == 0) {
#pragma unroll
    for (int nt = 0; nt < 4; ++nt)
#pragma unroll
      for (int r = 0; r < 8; ++r)
        q16[(size_t)(row0 + r + hi8) * Hn + nt * 16 + n] =
            (_Float16)(acc[nt][r] * 0.03125f);      // * C^-0.5
  } else if (mat == 1) {
#pragma unroll
    for (int nt = 0; nt < 4; ++nt)
#pragma unroll
      for (int r = 0; r < 8; ++r)
        k16[(size_t)(row0 + r + hi8) * Hn + nt * 16 + n] = (_Float16)acc[nt][r];
  } else {
    const int bb = row0 / Tn;
    const int t0 = row0 % Tn;
#pragma unroll
    for (int nt = 0; nt < 4; ++nt)
#pragma unroll
      for (int r = 0; r < 8; ++r)
        vTh[((size_t)bb * Hn + nt * 16 + n) * Tn + (t0 + r + hi8)] =
            (_Float16)acc[nt][r];
  }
}

// ---------------------------------------------------------------------------
// Kernel 2: flash attention. 4 waves/block; wave w owns the 16-row query tile
// at base + 16*w. K/V chunks (32 keys) are staged into LDS ONCE per block via
// async global->LDS copies and shared by all 4 waves. The chunk loop is
// uniform across the block; each wave masks elementwise for every chunk at or
// beyond its own diagonal (fully-masked chunks contribute exactly zero).
__global__ void __launch_bounds__(128)
attn_kernel(const _Float16* __restrict__ q16, const _Float16* __restrict__ k16,
            const _Float16* __restrict__ vTh, float* __restrict__ out) {
  __shared__ _Float16 kt[32][64];                   // K chunk: 32 keys x H
  __shared__ _Float16 vt[64][32];                   // V chunk: H x 32 keys (from vT)
  __shared__ _Float16 pbuf[4][16][32];              // per-wave P staging tile

  const int tid  = (int)threadIdx.x;
  const int wave = tid >> 5;
  const int b    = blockIdx.y;
  const int base = blockIdx.x * 64;
  const int m0   = base + wave * 16;                // query tile base (in batch)

  const _Float16* qb = q16 + (size_t)b * Tn * Hn;
  const _Float16* kb = k16 + (size_t)b * Tn * Hn;
  const _Float16* vb = vTh + (size_t)b * Hn * Tn;

  const int l   = laneid();
  const int n   = l & 15;
  const int hi8 = ((l >> 4) & 1) * 8;

  // Q A-fragments (K-dim = H = 64 -> two fragments), loaded once.
  v16h aq0 = load_fragA(qb, Hn, m0, 0);
  v16h aq1 = load_fragA(qb, Hn, m0, 32);

  float mrow[8], lrow[8];
  v8f o[4] = {};
#pragma unroll
  for (int r = 0; r < 8; ++r) { mrow[r] = -1e30f; lrow[r] = 0.0f; }

  const int nchunks = base / 32 + 2;                // covers keys 0 .. base+63
  const int diag_c  = m0 / 32;                      // this wave's diagonal chunk

  for (int c = 0; c < nchunks; ++c) {
    const int s0 = c * 32;

    // ---- stage K chunk (32x64) and V chunk (64x32) into LDS (async) ----
    __syncthreads();                                // prior reads done
#pragma unroll
    for (int j = 0; j < 2; ++j) {
      int seg = j * 128 + tid;                      // 256 x 16B segments each
      int kr = seg >> 3, kc = (seg & 7) * 8;        // K tile: 32 rows x 64 halves
      async_copy_b128(kb + (size_t)(s0 + kr) * Hn + kc, &kt[kr][kc]);
      int vr = seg >> 2, vc = (seg & 3) * 8;        // V tile: 64 rows x 32 halves
      async_copy_b128(vb + (size_t)vr * Tn + (s0 + vc), &vt[vr][vc]);
    }
    wait_async0();
    __syncthreads();                                // chunk visible to all waves

    // ---- S = Q K^T (pre-scaled): two 16x16 C-tiles, K operand from LDS ----
    v8f sc[2];
#pragma unroll
    for (int st = 0; st < 2; ++st) {
      v16h bk0 = load_fragB(&kt[0][0], Hn, st * 16, 0);
      v16h bk1 = load_fragB(&kt[0][0], Hn, st * 16, 32);
      v8f a = {};
      a = __builtin_amdgcn_wmma_f32_16x16x32_f16(false, aq0, false, bk0, (short)0, a, false, false);
      a = __builtin_amdgcn_wmma_f32_16x16x32_f16(false, aq1, false, bk1, (short)0, a, false, false);
      sc[st] = a;
    }

    // ---- causal mask: any chunk at/after this wave's diagonal ----
    if (c >= diag_c) {
#pragma unroll
      for (int st = 0; st < 2; ++st)
#pragma unroll
        for (int r = 0; r < 8; ++r) {
          int mg = m0 + r + hi8;
          int sg = s0 + st * 16 + n;
          if (sg > mg) sc[st][r] = -1e30f;
        }
    }

    // ---- online softmax (rows live across 16-lane groups, vgpr r) ----
#pragma unroll
    for (int r = 0; r < 8; ++r) {
      float mx = fmaxf(sc[0][r], sc[1][r]);
#pragma unroll
      for (int off = 8; off >= 1; off >>= 1)
        mx = fmaxf(mx, __shfl_xor(mx, off, 32));
      float mnew  = fmaxf(mrow[r], mx);
      float alpha = exp2f((mrow[r] - mnew) * kLog2e);
      mrow[r] = mnew;
      float p0 = exp2f((sc[0][r] - mnew) * kLog2e);
      float p1 = exp2f((sc[1][r] - mnew) * kLog2e);
      sc[0][r] = p0; sc[1][r] = p1;
      float rs = p0 + p1;
#pragma unroll
      for (int off = 8; off >= 1; off >>= 1)
        rs += __shfl_xor(rs, off, 32);
      lrow[r] = lrow[r] * alpha + rs;
#pragma unroll
      for (int nt = 0; nt < 4; ++nt) o[nt][r] *= alpha;
    }

    // ---- stage P (16x32 f16) through LDS: C-layout -> A-layout ----
#pragma unroll
    for (int r = 0; r < 8; ++r) {
      pbuf[wave][r + hi8][n]      = (_Float16)sc[0][r];
      pbuf[wave][r + hi8][16 + n] = (_Float16)sc[1][r];
    }
    asm volatile("s_wait_dscnt 0" ::: "memory");    // same-wave DS write->read
    v16h pa = load_fragA_lds(&pbuf[wave][0][0]);
    asm volatile("s_wait_dscnt 0" ::: "memory");    // reads done before next writes

    // ---- O += P V  (V operand from LDS) ----
#pragma unroll
    for (int nt = 0; nt < 4; ++nt) {
      v16h bv = load_fragB(&vt[0][0], 32, nt * 16, 0);
      o[nt] = __builtin_amdgcn_wmma_f32_16x16x32_f16(
          false, pa, false, bv, (short)0, o[nt], false, false);
    }
  }

  // ---- normalize and write fp32 output [B][T][H] ----
  float* ob = out + ((size_t)b * Tn + m0) * Hn;
#pragma unroll
  for (int nt = 0; nt < 4; ++nt)
#pragma unroll
    for (int r = 0; r < 8; ++r)
      ob[(size_t)(r + hi8) * Hn + nt * 16 + n] = o[nt][r] / lrow[r];
}

// ---------------------------------------------------------------------------
extern "C" void kernel_launch(void* const* d_in, const int* in_sizes, int n_in,
                              void* d_out, int out_size, void* d_ws, size_t ws_size,
                              hipStream_t stream) {
  const float* x  = (const float*)d_in[0];
  const float* Wq = (const float*)d_in[1];
  const float* Wk = (const float*)d_in[2];
  const float* Wv = (const float*)d_in[3];
  float* out = (float*)d_out;

  // Workspace carve-up (f16): xh | wt | q16 | k16 | vT   (~40 MB total)
  _Float16* xh  = (_Float16*)d_ws;
  _Float16* wt  = xh  + (size_t)Nrow * Cn;
  _Float16* q16 = wt  + (size_t)3 * Hn * Cn;
  _Float16* k16 = q16 + (size_t)Nrow * Hn;
  _Float16* vTh = k16 + (size_t)Nrow * Hn;

  const size_t nx = (size_t)Nrow * Cn;
  cvt_x_kernel<<<dim3((unsigned)((nx + 255) / 256)), dim3(256), 0, stream>>>(x, xh, nx);
  cvt_w_kernel<<<dim3((3 * Hn * Cn + 255) / 256), dim3(256), 0, stream>>>(Wq, Wk, Wv, wt);
  proj_kernel<<<dim3(Nrow / 16, 3), dim3(32), 0, stream>>>(xh, wt, q16, k16, vTh);
  attn_kernel<<<dim3(Tn / 64, Bn), dim3(128), 0, stream>>>(q16, k16, vTh, out);
}
